// CGNN_14087492731504
// MI455X (gfx1250) — matrix operand order repaired and checked by reading, not compiled, not run on hardware
//
#include <hip/hip_runtime.h>
#include <hip/hip_bf16.h>

// ---------------------------------------------------------------------------
// GaAN-style 2-layer GNN for MI455X (gfx1250, wave32).
//  - Dense GEMMs via v_wmma_f32_16x16x32_bf16 (bf16 operands, f32 accum)
//  - Sparse attention/aggregation via CSR, one wave32 per destination node
// ---------------------------------------------------------------------------

#define NND   50000
#define INDIM 512
#define HIDD  64
#define NHEAD 4
#define OUTD  64
#define NEGS  0.01f
#define ANEG  0.2f

typedef __attribute__((ext_vector_type(16))) __bf16        v16bf;
typedef __attribute__((ext_vector_type(8)))  float         v8f;
typedef __attribute__((ext_vector_type(8)))  unsigned short u16x8;
typedef __attribute__((ext_vector_type(16))) unsigned short u16x16;

union Frag16 {
  v16bf  v;
  u16x16 u;
  u16x8  h[2];
  unsigned short s[16];
};

__device__ __forceinline__ unsigned short f2bf(float f) {
  unsigned u = __builtin_bit_cast(unsigned, f);
  u += 0x7fffu + ((u >> 16) & 1u);            // round-to-nearest-even
  return (unsigned short)(u >> 16);
}
__device__ __forceinline__ float bf2f(unsigned short h) {
  return __builtin_bit_cast(float, ((unsigned)h) << 16);
}
__device__ __forceinline__ float lrelu(float x, float s) {
  return fmaxf(x, 0.f) + s * fminf(x, 0.f);
}

// ---------------------------------------------------------------------------
// CSR construction
// ---------------------------------------------------------------------------
__global__ void k_count(const int* __restrict__ dst, int* __restrict__ cnt, int E) {
  int i = blockIdx.x * 256 + threadIdx.x;
  if (i < E) atomicAdd(&cnt[dst[i]], 1);
}

// Exclusive scan of cnt[] (single block); rewrites cnt[] with start offsets
// (used as the scatter cursor) and fills rowptr[0..n].
__global__ void k_scan(int* __restrict__ cnt, int* __restrict__ rowptr, int n) {
  __shared__ int lds[1024];
  const int t = threadIdx.x;
  int carry = 0;
  for (int base = 0; base < n; base += 1024) {
    int v = (base + t < n) ? cnt[base + t] : 0;
    lds[t] = v;
    __syncthreads();
    for (int off = 1; off < 1024; off <<= 1) {
      int x = (t >= off) ? lds[t - off] : 0;
      __syncthreads();
      lds[t] += x;
      __syncthreads();
    }
    int excl = lds[t] - v;
    if (base + t < n) {
      rowptr[base + t] = carry + excl;
      cnt[base + t]    = carry + excl;   // cursor
    }
    int tot = lds[1023];
    __syncthreads();
    carry += tot;
  }
  if (t == 0) rowptr[n] = carry;
}

__global__ void k_scatter(const int* __restrict__ src, const int* __restrict__ dst,
                          int* __restrict__ cursor, int* __restrict__ csrc, int E) {
  int i = blockIdx.x * 256 + threadIdx.x;
  if (i < E) {
    int slot = atomicAdd(&cursor[dst[i]], 1);
    csrc[slot] = src[i];
  }
}

// ---------------------------------------------------------------------------
// Weight packing: f32 [K,Nc] -> bf16 in WMMA B-fragment order.
// Packed index = ((kb*(Nc/16) + nt)*32 + lane)*16 + j
//   src K = kb*32 + 8*(lane>>4) + (j&7) + 16*(j>>3) ; src col = nt*16 + (lane&15)
// mode 0: row-major W[K,Nc];  mode 1: Wv layout [h][k][o] with col = h*64+o
// ---------------------------------------------------------------------------
__global__ void k_pack(const float* __restrict__ W, unsigned short* __restrict__ out,
                       int K, int Nc, int mode) {
  int idx = blockIdx.x * 256 + threadIdx.x;
  int total = K * Nc;
  if (idx >= total) return;
  int perKb = Nc * 32;
  int kb = idx / perKb;
  int r  = idx % perKb;
  int nt = r >> 9;          // / (32*16)
  int r2 = r & 511;
  int lane = r2 >> 4;
  int j    = r2 & 15;
  int sk = kb * 32 + ((lane >> 4) << 3) + (j & 7) + ((j >> 3) << 4);
  int sc = nt * 16 + (lane & 15);
  float v = (mode == 0) ? W[(size_t)sk * Nc + sc]
                        : W[((sc >> 6) << 12) + sk * 64 + (sc & 63)];
  out[idx] = f2bf(v);
}

// AB[l][k][j] = sum_d We[k][d] * a_{l or r}[layer l][h=j&3][d]   (j<4 -> a_l)
__global__ void k_ab(const float* __restrict__ We, const float* __restrict__ a_l,
                     const float* __restrict__ a_r, float* __restrict__ AB) {
  int idx = blockIdx.x * 256 + threadIdx.x;
  if (idx >= 1024) return;
  int l = idx >> 9;
  int r = idx & 511;
  int k = r >> 3;
  int j = r & 7;
  const float* av = ((j < 4) ? a_l : a_r) + (size_t)(l * 4 + (j & 3)) * 64;
  const float* wr = We + (size_t)k * 64;
  float acc = 0.f;
  #pragma unroll 8
  for (int d = 0; d < 64; ++d) acc += wr[d] * av[d];
  AB[l * 512 + k * 8 + j] = acc;
}

// eler[n][j] = sum_k curbf[n][k] * AB[k][j]   (cols 0..3 = el, 4..7 = er)
__global__ void k_eler(const unsigned short* __restrict__ curbf,
                       const float* __restrict__ AB,
                       float* __restrict__ eler, int n) {
  int idx = blockIdx.x * 256 + threadIdx.x;
  if (idx >= n * 8) return;
  int node = idx >> 3, j = idx & 7;
  const unsigned short* cr = curbf + (size_t)node * 64;
  float acc = 0.f;
  #pragma unroll 8
  for (int k = 0; k < 64; ++k) acc += bf2f(cr[k]) * AB[k * 8 + j];
  eler[idx] = acc;
}

// ---------------------------------------------------------------------------
// WMMA GEMM: C[M,Nc] = A[M,K] x Bpacked[K,Nc], A f32 (convert) or bf16.
// Block = 256 thr = 8 waves; each wave -> 16 rows x 64 cols (4 wmma tiles).
// grid = (ceil(M/128), Nc/64). Optional bias, plus up to two bf16 copies.
// ---------------------------------------------------------------------------
template <bool AF32>
__global__ __launch_bounds__(256) void k_gemm(
    const void* __restrict__ Aptr, const unsigned short* __restrict__ Bp,
    int M, int K, int Nc,
    float* __restrict__ Cf, const float* __restrict__ bias,
    unsigned short* __restrict__ Cb, int ldcb,
    unsigned short* __restrict__ Cb2, int ldcb2) {
  const int lane = threadIdx.x & 31;
  const int wid  = threadIdx.x >> 5;
  const int tileM = (blockIdx.x * 8 + wid) * 16;
  if (tileM >= M) return;
  const int tileN = blockIdx.y * 64;
  int rA = tileM + (lane & 15);
  if (rA >= M) rA = M - 1;                   // clamp (stores are masked)
  const int khalf = lane >> 4;
  const int ntiles = Nc >> 4;

  v8f acc[4];
  #pragma unroll
  for (int i = 0; i < 4; ++i)
    #pragma unroll
    for (int j = 0; j < 8; ++j) acc[i][j] = 0.f;

  const int nkb = K >> 5;
  for (int kb = 0; kb < nkb; ++kb) {
    Frag16 a;
    if constexpr (AF32) {
      const float* ap = (const float*)Aptr + (size_t)rA * K + kb * 32 + khalf * 8;
      #pragma unroll
      for (int j = 0; j < 8; ++j) {
        a.s[j]     = f2bf(ap[j]);
        a.s[8 + j] = f2bf(ap[16 + j]);
      }
    } else {
      const unsigned short* ap =
          (const unsigned short*)Aptr + (size_t)rA * K + kb * 32 + khalf * 8;
      a.h[0] = *(const u16x8*)ap;
      a.h[1] = *(const u16x8*)(ap + 16);
    }
    #pragma unroll
    for (int nt = 0; nt < 4; ++nt) {
      Frag16 b;
      b.u = *(const u16x16*)(Bp +
            (((size_t)(kb * ntiles + (blockIdx.y * 4 + nt)) * 32 + lane) << 4));
      acc[nt] = __builtin_amdgcn_wmma_f32_16x16x32_bf16(
          false, a.v, false, b.v, (short)0, acc[nt], false, false);
    }
  }

  const int rowBase = tileM + ((lane >> 4) << 3);
  const int colLo   = lane & 15;
  #pragma unroll
  for (int nt = 0; nt < 4; ++nt) {
    const int c = tileN + nt * 16 + colLo;
    const float bv = bias ? bias[c] : 0.f;
    #pragma unroll
    for (int v = 0; v < 8; ++v) {
      const int row = rowBase + v;
      if (row < M) {
        const float val = acc[nt][v] + bv;
        if (Cf)  Cf[(size_t)row * Nc + c] = val;
        if (Cb)  Cb[(size_t)row * ldcb + c]  = f2bf(val);
        if (Cb2) Cb2[(size_t)row * ldcb2 + c] = f2bf(val);
      }
    }
  }
}

// ---------------------------------------------------------------------------
// Per-node aggregation: one wave32 per destination node.
// pass1/2: segment softmax stats (edges distributed across lanes)
// pass3  : serial edge loop, lanes cover 256 (h,o) accumulators + mx/mn
// epilogue: GaAN gate + merge + leaky, fused.
// ---------------------------------------------------------------------------
__global__ __launch_bounds__(256) void k_agg(
    const int* __restrict__ rowptr, const int* __restrict__ csrc,
    const float* __restrict__ eler, const float* __restrict__ vals,
    const float* __restrict__ pooled, const float* __restrict__ curin,
    const float* __restrict__ Wgate, const float* __restrict__ bgate,
    float* __restrict__ curout, unsigned short* __restrict__ curbf,
    unsigned short* __restrict__ hbf, int layer, int nNodes) {
  __shared__ float sWg[768];
  __shared__ float sBg[4];
  for (int i = threadIdx.x; i < 768; i += 256) sWg[i] = Wgate[i];
  if (threadIdx.x < 4) sBg[threadIdx.x] = bgate[threadIdx.x];
  __syncthreads();

  const int lane = threadIdx.x & 31;
  const int n = blockIdx.x * 8 + (threadIdx.x >> 5);
  if (n >= nNodes) return;
  const int start = rowptr[n], end = rowptr[n + 1];
  const float4 ern = *(const float4*)(eler + (size_t)n * 8 + 4);

  // ---- pass 1: per-head max ----
  float m0 = -3.0e38f, m1 = -3.0e38f, m2 = -3.0e38f, m3 = -3.0e38f;
  for (int j = start + lane; j < end; j += 32) {
    const int s = csrc[j];
    const float4 el = *(const float4*)(eler + (size_t)s * 8);
    m0 = fmaxf(m0, lrelu(el.x + ern.x, ANEG));
    m1 = fmaxf(m1, lrelu(el.y + ern.y, ANEG));
    m2 = fmaxf(m2, lrelu(el.z + ern.z, ANEG));
    m3 = fmaxf(m3, lrelu(el.w + ern.w, ANEG));
  }
  #pragma unroll
  for (int off = 16; off; off >>= 1) {
    m0 = fmaxf(m0, __shfl_xor(m0, off, 32));
    m1 = fmaxf(m1, __shfl_xor(m1, off, 32));
    m2 = fmaxf(m2, __shfl_xor(m2, off, 32));
    m3 = fmaxf(m3, __shfl_xor(m3, off, 32));
  }

  // ---- pass 2: exp-sum ----
  float s0 = 0.f, s1 = 0.f, s2 = 0.f, s3 = 0.f;
  for (int j = start + lane; j < end; j += 32) {
    const int s = csrc[j];
    const float4 el = *(const float4*)(eler + (size_t)s * 8);
    s0 += __expf(lrelu(el.x + ern.x, ANEG) - m0);
    s1 += __expf(lrelu(el.y + ern.y, ANEG) - m1);
    s2 += __expf(lrelu(el.z + ern.z, ANEG) - m2);
    s3 += __expf(lrelu(el.w + ern.w, ANEG) - m3);
  }
  #pragma unroll
  for (int off = 16; off; off >>= 1) {
    s0 += __shfl_xor(s0, off, 32);
    s1 += __shfl_xor(s1, off, 32);
    s2 += __shfl_xor(s2, off, 32);
    s3 += __shfl_xor(s3, off, 32);
  }
  const float d0 = 1.f / s0, d1 = 1.f / s1, d2 = 1.f / s2, d3 = 1.f / s3;

  // ---- pass 3: weighted aggregation + pool max + mean ----
  float agg[8] = {0.f, 0.f, 0.f, 0.f, 0.f, 0.f, 0.f, 0.f};
  float mx0 = -3.0e38f, mx1 = -3.0e38f, mn0 = 0.f, mn1 = 0.f;
  for (int j = start; j < end; ++j) {
    const int s = csrc[j];
    const float4 el = *(const float4*)(eler + (size_t)s * 8);
    const float ah[4] = {
        __expf(lrelu(el.x + ern.x, ANEG) - m0) * d0,
        __expf(lrelu(el.y + ern.y, ANEG) - m1) * d1,
        __expf(lrelu(el.z + ern.z, ANEG) - m2) * d2,
        __expf(lrelu(el.w + ern.w, ANEG) - m3) * d3};
    const float* vr = vals + (size_t)s * 256;
    #pragma unroll
    for (int k = 0; k < 8; ++k) agg[k] += ah[k >> 1] * vr[lane + 32 * k];
    const float* pr = pooled + (size_t)s * 64;
    mx0 = fmaxf(mx0, pr[lane]);
    mx1 = fmaxf(mx1, pr[lane + 32]);
    const float* cr = curin + (size_t)s * 64;
    mn0 += cr[lane];
    mn1 += cr[lane + 32];
  }
  const float invd = 1.f / (float)(end - start);
  mn0 *= invd;
  mn1 *= invd;

  // ---- gate: sigmoid([cur, mx, mn] @ Wgate + bgate) ----
  const float cn0 = curin[(size_t)n * 64 + lane];
  const float cn1 = curin[(size_t)n * 64 + lane + 32];
  const float gi[6] = {cn0, cn1, mx0, mx1, mn0, mn1};
  float g0 = 0.f, g1 = 0.f, g2 = 0.f, g3 = 0.f;
  #pragma unroll
  for (int t = 0; t < 6; ++t) {
    const int f = lane + 32 * t;
    const float x = gi[t];
    g0 += x * sWg[f * 4 + 0];
    g1 += x * sWg[f * 4 + 1];
    g2 += x * sWg[f * 4 + 2];
    g3 += x * sWg[f * 4 + 3];
  }
  #pragma unroll
  for (int off = 16; off; off >>= 1) {
    g0 += __shfl_xor(g0, off, 32);
    g1 += __shfl_xor(g1, off, 32);
    g2 += __shfl_xor(g2, off, 32);
    g3 += __shfl_xor(g3, off, 32);
  }
  g0 = 1.f / (1.f + __expf(-(g0 + sBg[0])));
  g1 = 1.f / (1.f + __expf(-(g1 + sBg[1])));
  g2 = 1.f / (1.f + __expf(-(g2 + sBg[2])));
  g3 = 1.f / (1.f + __expf(-(g3 + sBg[3])));

  if (layer == 0) {
    // mean over heads: lane holds o=lane (even k) and o=lane+32 (odd k)
    float mA = 0.25f * (g0 * agg[0] + g1 * agg[2] + g2 * agg[4] + g3 * agg[6]);
    float mB = 0.25f * (g0 * agg[1] + g1 * agg[3] + g2 * agg[5] + g3 * agg[7]);
    mA = lrelu(mA, NEGS);
    mB = lrelu(mB, NEGS);
    curout[(size_t)n * 64 + lane]      = mA;
    curout[(size_t)n * 64 + lane + 32] = mB;
    curbf[(size_t)n * 64 + lane]       = f2bf(mA);
    curbf[(size_t)n * 64 + lane + 32]  = f2bf(mB);
  } else {
    const float gh[4] = {g0, g1, g2, g3};
    #pragma unroll
    for (int k = 0; k < 8; ++k) {
      const float v = lrelu(gh[k >> 1] * agg[k], NEGS);
      hbf[(size_t)n * 320 + 64 + lane + 32 * k] = f2bf(v);  // h = [pf | cur]
    }
  }
}

// ---------------------------------------------------------------------------
extern "C" void kernel_launch(void* const* d_in, const int* in_sizes, int n_in,
                              void* d_out, int out_size, void* d_ws, size_t ws_size,
                              hipStream_t stream) {
  const float* ft    = (const float*)d_in[0];
  const float* Wp    = (const float*)d_in[1];
  const float* We    = (const float*)d_in[2];
  const float* a_l   = (const float*)d_in[3];
  const float* a_r   = (const float*)d_in[4];
  const float* Wv    = (const float*)d_in[5];
  const float* Wgp   = (const float*)d_in[6];
  const float* Wgate = (const float*)d_in[7];
  const float* bgate = (const float*)d_in[8];
  const float* Wtr   = (const float*)d_in[9];
  const float* btr   = (const float*)d_in[10];
  const int*   src   = (const int*)d_in[11];
  const int*   dst   = (const int*)d_in[12];
  const int E = in_sizes[11];
  const int N = NND;

  char* base = (char*)d_ws;
  size_t off = 0;
  auto take = [&](size_t bytes) -> char* {
    char* p = base + off;
    off = (off + bytes + 255) & ~(size_t)255;
    return p;
  };
  int*   rowptr = (int*)take((size_t)(N + 1) * 4);
  int*   cursor = (int*)take((size_t)N * 4);
  int*   csrc   = (int*)take((size_t)E * 4);
  float* pf32   = (float*)take((size_t)N * 64 * 4);
  float* cur1   = (float*)take((size_t)N * 64 * 4);
  unsigned short* curbf = (unsigned short*)take((size_t)N * 64 * 2);
  float* eler   = (float*)take((size_t)N * 8 * 4);
  float* vals   = (float*)take((size_t)N * 256 * 4);
  float* pooled = (float*)take((size_t)N * 64 * 4);
  unsigned short* hbf  = (unsigned short*)take((size_t)N * 320 * 2);
  unsigned short* wpB  = (unsigned short*)take((size_t)512 * 64 * 2);
  unsigned short* wvB  = (unsigned short*)take((size_t)2 * 64 * 256 * 2);
  unsigned short* wgpB = (unsigned short*)take((size_t)2 * 64 * 64 * 2);
  unsigned short* wtrB = (unsigned short*)take((size_t)320 * 64 * 2);
  float* ABw = (float*)take((size_t)2 * 64 * 8 * 4);
  if (off > ws_size) return;  // workspace too small; bail safely

  // ---- CSR build ----
  hipMemsetAsync(cursor, 0, (size_t)N * 4, stream);
  k_count<<<(E + 255) / 256, 256, 0, stream>>>(dst, cursor, E);
  k_scan<<<1, 1024, 0, stream>>>(cursor, rowptr, N);
  k_scatter<<<(E + 255) / 256, 256, 0, stream>>>(src, dst, cursor, csrc, E);

  // ---- weight packing (bf16, WMMA B-fragment order) ----
  k_pack<<<(512 * 64 + 255) / 256, 256, 0, stream>>>(Wp, wpB, 512, 64, 0);
  k_pack<<<(64 * 256 + 255) / 256, 256, 0, stream>>>(Wv, wvB, 64, 256, 1);
  k_pack<<<(64 * 256 + 255) / 256, 256, 0, stream>>>(Wv + 16384, wvB + 16384, 64, 256, 1);
  k_pack<<<(64 * 64 + 255) / 256, 256, 0, stream>>>(Wgp, wgpB, 64, 64, 0);
  k_pack<<<(64 * 64 + 255) / 256, 256, 0, stream>>>(Wgp + 4096, wgpB + 4096, 64, 64, 0);
  k_pack<<<(320 * 64 + 255) / 256, 256, 0, stream>>>(Wtr, wtrB, 320, 64, 0);
  k_ab<<<4, 256, 0, stream>>>(We, a_l, a_r, ABw);

  dim3 g1((N + 127) / 128, 1), g4((N + 127) / 128, 4);

  // proj_feat = ft @ Wp  (f32 A -> bf16 on the fly); also seeds h[:,0:64] (bf16)
  k_gemm<true><<<g1, 256, 0, stream>>>(ft, wpB, N, 512, 64,
                                       pf32, nullptr, curbf, 64, hbf, 320);

  for (int l = 0; l < 2; ++l) {
    const float* curin = l ? cur1 : pf32;
    k_eler<<<((N * 8) + 255) / 256, 256, 0, stream>>>(curbf, ABw + l * 512, eler, N);
    k_gemm<false><<<g4, 256, 0, stream>>>(curbf, wvB + l * 16384, N, 64, 256,
                                          vals, nullptr, nullptr, 0, nullptr, 0);
    k_gemm<false><<<g1, 256, 0, stream>>>(curbf, wgpB + l * 4096, N, 64, 64,
                                          pooled, nullptr, nullptr, 0, nullptr, 0);
    k_agg<<<(N + 7) / 8, 256, 0, stream>>>(rowptr, csrc, eler, vals, pooled, curin,
                                           Wgate + l * 768, bgate + l * 4,
                                           cur1, curbf, hbf, l, N);
  }

  // out = [proj_feat | cur] @ Wtran + btran
  k_gemm<false><<<g1, 256, 0, stream>>>(hbf, wtrB, N, 320, 64,
                                        (float*)d_out, btr, nullptr, 0, nullptr, 0);
}
